// EGNNDenoiser_80444737454135
// MI455X (gfx1250) — matrix operand order repaired and checked by reading, not compiled, not run on hardware
//
#include <hip/hip_runtime.h>

// ---------------------------------------------------------------------------
// EGNN denoiser for MI455X (gfx1250, wave32).
//  - All GEMMs: V_WMMA_F32_16X16X32_BF16 (f32 accumulate)
//  - h activations stored bf16 (A operands load with zero conversion ALU)
//  - Weight B-operand register images prebuilt in global ws, staged to LDS
//    via GLOBAL_LOAD_ASYNC_TO_LDS_B128 (ASYNCcnt)
//  - Native __bf16 converts (v_cvt_pk_bf16_f32) instead of manual rounding
//  - r-outer epilogues: per-row base pointers so store/atomic offsets fold
//    into instruction immediates
// ---------------------------------------------------------------------------

#define NND   10000      // nodes
#define NDIM  64         // node_dim
#define HD    128        // hidden
#define NL    4          // layers
#define KNN   16         // k
#define TDIM  16         // time dim
#define NTILE (NND / 16) // 625 16-row tiles (exact)

typedef __attribute__((ext_vector_type(16))) __bf16 v16bf;
typedef __attribute__((ext_vector_type(8)))  float  v8f;

union V16BF { v16bf v; unsigned int u[8]; };

__device__ __forceinline__ unsigned short f2bf(float f) {
  union { __bf16 h; unsigned short u; } cv;
  cv.h = (__bf16)f;                          // native cvt, RTE
  return cv.u;
}
__device__ __forceinline__ float bf2f(unsigned short b) {
  return __uint_as_float(((unsigned int)b) << 16);
}
__device__ __forceinline__ unsigned int pack2(float f0, float f1) {
  union { __bf16 h[2]; unsigned int u; } cv;
  cv.h[0] = (__bf16)f0; cv.h[1] = (__bf16)f1;  // fuses to v_cvt_pk_bf16_f32
  return cv.u;
}
__device__ __forceinline__ float silu(float x) {
  return x / (1.0f + __expf(-x));
}
__device__ __forceinline__ v8f v8zero() {
  v8f z = {0.f, 0.f, 0.f, 0.f, 0.f, 0.f, 0.f, 0.f};
  return z;
}
__device__ __forceinline__ v8f wmma_bf16(const V16BF& a, const V16BF& b, v8f c) {
  return __builtin_amdgcn_wmma_f32_16x16x32_bf16(
      false, a.v, false, b.v, (short)0, c, false, false);
}
__device__ __forceinline__ void fatomic_add(float* p, float v) {
  __hip_atomic_fetch_add(p, v, __ATOMIC_RELAXED, __HIP_MEMORY_SCOPE_AGENT);
}

// ---- A operand (16x32 bf16, MxK) ------------------------------------------
// lane l: M=l&15; g=l>>4. VGPR v<4: K=8g+2v{,+1}; v>=4: K=16+8g+2(v-4){,+1}
__device__ __forceinline__ V16BF load_A_bf16(const unsigned short* row, int kbase, int lane) {
  V16BF a; int g8 = (lane >> 4) << 3;
  const unsigned int* rp = (const unsigned int*)row;
#pragma unroll
  for (int v = 0; v < 8; v++) {
    int k = kbase + ((v < 4) ? 0 : 16) + g8 + ((v & 3) << 1);
    a.u[v] = rp[k >> 1];                      // bf16 pair, no conversion ALU
  }
  return a;
}
__device__ __forceinline__ V16BF load_A_f32_scaled(const float* row, float sc, int kbase, int lane) {
  V16BF a; int g8 = (lane >> 4) << 3;
#pragma unroll
  for (int v = 0; v < 8; v++) {
    int k = kbase + ((v < 4) ? 0 : 16) + g8 + ((v & 3) << 1);
    a.u[v] = pack2(row[k] * sc, row[k + 1] * sc);
  }
  return a;
}
// A from row-major 16x128 bf16 LDS tile
__device__ __forceinline__ V16BF load_A_lds(const unsigned short* buf, int kbase, int lane) {
  V16BF a; int g8 = (lane >> 4) << 3;
  const unsigned int* rp = (const unsigned int*)buf + (lane & 15) * 64;
#pragma unroll
  for (int v = 0; v < 8; v++) {
    int k = kbase + ((v < 4) ? 0 : 16) + g8 + ((v & 3) << 1);
    a.u[v] = rp[k >> 1];
  }
  return a;
}

// ---- B operand images ------------------------------------------------------
// B (32x16 bf16, KxN): lane = 16*((k>>3)&1) + (n&15); VGPR v=4*(k>>4)+((k&7)>>1); half=k&1.
// Image: block b = (n>>4)*ksteps + (k>>5), 256 dwords per block, lane-major.
__global__ void build_image_kernel(const float* __restrict__ W, int Kact, int Kpad,
                                   unsigned int* __restrict__ img) {
  int ksteps = Kpad >> 5;
  int tot = (Kpad >> 1) * HD;
  for (int pI = blockIdx.x * blockDim.x + threadIdx.x; pI < tot;
       pI += gridDim.x * blockDim.x) {
    int kp = pI >> 7, n = pI & 127, k = kp << 1;
    float f0 = (k < Kact) ? W[k * HD + n] : 0.f;
    float f1 = (k + 1 < Kact) ? W[(k + 1) * HD + n] : 0.f;
    int t = k >> 5, kk = k & 31;
    int g = (kk >> 3) & 1, v = ((kk >> 4) << 2) + ((kk & 7) >> 1);
    int lane = g * 16 + (n & 15);
    int j = n >> 4;
    img[(j * ksteps + t) * 256 + lane * 8 + v] = pack2(f0, f1);
  }
}
__device__ __forceinline__ V16BF load_B(const unsigned int* img, int blk, int lane) {
  V16BF b;
  const unsigned int* p = img + blk * 256 + lane * 8;
#pragma unroll
  for (int i = 0; i < 8; i++) b.u[i] = p[i];
  return b;
}

// ---- Stage a prebuilt image from global into LDS ---------------------------
typedef int v4i __attribute__((vector_size(4 * sizeof(int))));
typedef __attribute__((address_space(1))) v4i GlobalV4i;   // AS1: "__device__"
typedef __attribute__((address_space(3))) v4i LocalV4i;    // AS3: "__shared__"

__device__ __forceinline__ void stage_copy(unsigned int* ldst,
                                           const unsigned int* gsrc, int ndwords) {
  int nvec = ndwords >> 2;
#if __has_builtin(__builtin_amdgcn_global_load_async_to_lds_b128)
  for (int i = threadIdx.x; i < nvec; i += blockDim.x) {
    __builtin_amdgcn_global_load_async_to_lds_b128(
        (GlobalV4i*)(gsrc + 4 * i), (LocalV4i*)(ldst + 4 * i), 0, 0);
  }
#if __has_builtin(__builtin_amdgcn_s_wait_asynccnt)
  __builtin_amdgcn_s_wait_asynccnt(0);
#else
  asm volatile("s_wait_asynccnt 0" ::: "memory");
#endif
#else
  const uint4* s4 = (const uint4*)gsrc;
  uint4* d4 = (uint4*)ldst;
  for (int i = threadIdx.x; i < nvec; i += blockDim.x) d4[i] = s4[i];
#endif
}

// ===========================================================================
// kNN: brute force with LDS position tiles + branchless register top-16
// ===========================================================================
#define KNN_TJ 2048
__global__ void knn_kernel(const float* __restrict__ pos, int* __restrict__ nbr) {
  __shared__ float sp[KNN_TJ * 3];
  int i = blockIdx.x * blockDim.x + threadIdx.x;
  float px = 0.f, py = 0.f, pz = 0.f;
  if (i < NND) { px = pos[i * 3]; py = pos[i * 3 + 1]; pz = pos[i * 3 + 2]; }
  float bd[KNN]; int bi[KNN];
#pragma unroll
  for (int q = 0; q < KNN; q++) { bd[q] = 3.0e38f; bi[q] = 0; }
  for (int jb = 0; jb < NND; jb += KNN_TJ) {
    int cnt = (NND - jb < KNN_TJ) ? (NND - jb) : KNN_TJ;
    for (int c = threadIdx.x; c < cnt * 3; c += blockDim.x) sp[c] = pos[jb * 3 + c];
    __syncthreads();
    if (i < NND) {
      for (int j = 0; j < cnt; j++) {
        float dx = px - sp[j * 3], dy = py - sp[j * 3 + 1], dz = pz - sp[j * 3 + 2];
        float d2 = dx * dx + dy * dy + dz * dz;
        int gj = jb + j;
        if (gj != i && d2 < bd[KNN - 1]) {
          bool placed = false;
#pragma unroll
          for (int q = KNN - 1; q >= 1; q--) {
            bool shift = (!placed) && (d2 < bd[q - 1]);
            if (shift) { bd[q] = bd[q - 1]; bi[q] = bi[q - 1]; }
            else if (!placed) { bd[q] = d2; bi[q] = gj; placed = true; }
          }
          if (!placed) { bd[0] = d2; bi[0] = gj; }
        }
      }
    }
    __syncthreads();
  }
  if (i < NND) {
#pragma unroll
    for (int q = 0; q < KNN; q++) nbr[i * KNN + q] = bi[q];
  }
}

__global__ void zero_kernel(float* __restrict__ p, int n) {
  for (int i = blockIdx.x * blockDim.x + threadIdx.x; i < n; i += gridDim.x * blockDim.x)
    p[i] = 0.f;
}
__global__ void deg_kernel(const int* __restrict__ nbr, float* __restrict__ deg) {
  int e = blockIdx.x * blockDim.x + threadIdx.x;
  if (e < NND * KNN) fatomic_add(&deg[nbr[e]], 1.0f);
}

// ===========================================================================
// Input projection: h0 = [x | s | temb] @ proj_w + proj_b   (K padded 81->96)
// ===========================================================================
__device__ __forceinline__ float proj_in(const float* x, const float* s,
                                         const float* te, int node, int c) {
  if (c < NDIM) return x[node * NDIM + c];
  if (c == NDIM) return s[node];
  if (c < NDIM + 1 + TDIM) return te[c - (NDIM + 1)];
  return 0.f;
}

__global__ void __launch_bounds__(128) proj_kernel(
    const float* __restrict__ x, const float* __restrict__ s,
    const float* __restrict__ tptr, const unsigned int* __restrict__ imgW,
    const float* __restrict__ pb, unsigned short* __restrict__ hout) {
  extern __shared__ char smem[];
  unsigned int* Wimg = (unsigned int*)smem;      // 8 j * 3 ksteps * 256 dw
  float* bb = (float*)(Wimg + 8 * 3 * 256);
  float* te = bb + HD;                           // 16-float time embedding
  stage_copy(Wimg, imgW, 8 * 3 * 256);
  float tval = tptr[0];
  for (int c = threadIdx.x; c < HD; c += blockDim.x) bb[c] = pb[c];
  if (threadIdx.x < TDIM) {
    int i = threadIdx.x;
    int ii = (i < 8) ? i : (i - 8);
    float f = __expf(-4.0f * (float)ii / 7.0f);
    float a = tval * f;
    te[i] = (i < 8) ? __sinf(a) : __cosf(a);
  }
  __syncthreads();
  int wid = threadIdx.x >> 5, lane = threadIdx.x & 31;
  int q = lane & 15, hi = lane >> 4;
  int nwaves = gridDim.x * (blockDim.x >> 5);
  for (int tile = blockIdx.x * (blockDim.x >> 5) + wid; tile < NTILE; tile += nwaves) {
    int node = tile * 16 + q;
    v8f acc[8];
#pragma unroll
    for (int j = 0; j < 8; j++) acc[j] = v8zero();
#pragma unroll
    for (int t = 0; t < 3; t++) {
      V16BF A; int g8 = hi << 3;
#pragma unroll
      for (int v = 0; v < 8; v++) {
        int k = 32 * t + ((v < 4) ? 0 : 16) + g8 + ((v & 3) << 1);
        A.u[v] = pack2(proj_in(x, s, te, node, k), proj_in(x, s, te, node, k + 1));
      }
#pragma unroll
      for (int j = 0; j < 8; j++) {
        V16BF B = load_B(Wimg, j * 3 + t, lane);
        acc[j] = wmma_bf16(A, B, acc[j]);
      }
    }
    float bv[8];
#pragma unroll
    for (int j = 0; j < 8; j++) bv[j] = bb[q + 16 * j];
#pragma unroll
    for (int r = 0; r < 8; r++) {              // r-outer: one base ptr per row
      int M = r + 8 * hi;
      unsigned short* op = hout + (size_t)(tile * 16 + M) * HD + q;
#pragma unroll
      for (int j = 0; j < 8; j++) op[16 * j] = f2bf(acc[j][r] + bv[j]);
    }
  }
}

// ===========================================================================
// Edge kernel: one wave per source node (its 16 edges = one 16-row WMMA tile)
//   m = silu(silu([h[dst]|h[src]|r2]@W1+b1)@W2+b2) * s[src]
//   msum[dst] += m ; cu[dst] += (m.cw+cb)*dir     (global f32 atomics)
// ===========================================================================
#define EDGE_PW (16 * HD * 2 + 16 * 4 + 16 * 4 + 3 * 16 * 4) // m1buf,r2a,dsta,dir
__global__ void __launch_bounds__(128) edge_kernel(
    const unsigned short* __restrict__ h, const float* __restrict__ p,
    const float* __restrict__ s, const int* __restrict__ nbr,
    const unsigned int* __restrict__ imgW1, const unsigned int* __restrict__ imgW2,
    const float* __restrict__ ew1, const float* __restrict__ eb1,
    const float* __restrict__ eb2, const float* __restrict__ cw,
    const float* __restrict__ cb, float* __restrict__ msum, float* __restrict__ cu) {
  extern __shared__ char smem[];
  unsigned int* W1img = (unsigned int*)smem;   // 64 blocks * 256 dw
  unsigned int* W2img = W1img + 64 * 256;      // 32 blocks * 256 dw
  float* w1r = (float*)(W2img + 32 * 256);
  float* b1 = w1r + HD;
  float* b2 = b1 + HD;
  float* cwl = b2 + HD;
  char* wavearea = (char*)(cwl + HD);
  int wid = threadIdx.x >> 5, lane = threadIdx.x & 31;
  char* wa = wavearea + wid * EDGE_PW;
  unsigned short* m1buf = (unsigned short*)wa;   // 16 x 128 bf16
  float* r2a = (float*)(m1buf + 16 * HD);
  int* dsta = (int*)(r2a + 16);
  float* dxa = (float*)(dsta + 16);
  float* dya = dxa + 16;
  float* dza = dya + 16;

  stage_copy(W1img, imgW1, 64 * 256);
  stage_copy(W2img, imgW2, 32 * 256);
  for (int c = threadIdx.x; c < HD; c += blockDim.x) {
    w1r[c] = ew1[256 * HD + c];                // r2 row of ew1
    b1[c] = eb1[c]; b2[c] = eb2[c]; cwl[c] = cw[c];
  }
  __syncthreads();

  int q = lane & 15, hi = lane >> 4;
  float cb0 = cb[0];
  int nwaves = gridDim.x * (blockDim.x >> 5);
  for (int i = blockIdx.x * (blockDim.x >> 5) + wid; i < NND; i += nwaves) {
    int dq = nbr[i * KNN + q];
    float psx = p[i * 3], psy = p[i * 3 + 1], psz = p[i * 3 + 2];
    float dx = p[dq * 3] - psx, dy = p[dq * 3 + 1] - psy, dz = p[dq * 3 + 2] - psz;
    float r2 = dx * dx + dy * dy + dz * dz;
    float rinv = 1.0f / sqrtf(r2 + 1e-8f);
    if (lane < 16) {
      r2a[q] = r2; dsta[q] = dq;
      dxa[q] = dx * rinv; dya[q] = dy * rinv; dza[q] = dz * rinv;
    }
    __builtin_amdgcn_wave_barrier();

    v8f acc[8];
#pragma unroll
    for (int j = 0; j < 8; j++) acc[j] = v8zero();
    const unsigned short* hsrc = h + (size_t)i * HD;
    const unsigned short* hdst = h + (size_t)dq * HD;  // lane row = lane&15 -> own dq
#pragma unroll
    for (int t = 0; t < 4; t++) {              // e_in rows 0..127 : h[dst]
      V16BF A = load_A_bf16(hdst, 32 * t, lane);
#pragma unroll
      for (int j = 0; j < 8; j++) {
        V16BF B = load_B(W1img, j * 8 + t, lane);
        acc[j] = wmma_bf16(A, B, acc[j]);
      }
    }
#pragma unroll
    for (int t = 0; t < 4; t++) {              // e_in rows 128..255 : h[src]
      V16BF A = load_A_bf16(hsrc, 32 * t, lane);
#pragma unroll
      for (int j = 0; j < 8; j++) {
        V16BF B = load_B(W1img, j * 8 + 4 + t, lane);
        acc[j] = wmma_bf16(A, B, acc[j]);
      }
    }
    // epilogue 1: + r2*w1r + b1, silu -> LDS bf16 (A-layout reload next)
    {
      float wnv[8], b1v[8];
#pragma unroll
      for (int j = 0; j < 8; j++) { wnv[j] = w1r[q + 16 * j]; b1v[j] = b1[q + 16 * j]; }
#pragma unroll
      for (int r = 0; r < 8; r++) {
        int M = r + 8 * hi;
        float r2M = r2a[M];
        unsigned short* mp = m1buf + M * HD + q;
#pragma unroll
        for (int j = 0; j < 8; j++)
          mp[16 * j] = f2bf(silu(acc[j][r] + r2M * wnv[j] + b1v[j]));
      }
    }
    __builtin_amdgcn_wave_barrier();
    // GEMM2
#pragma unroll
    for (int j = 0; j < 8; j++) acc[j] = v8zero();
#pragma unroll
    for (int t = 0; t < 4; t++) {
      V16BF A = load_A_lds(m1buf, 32 * t, lane);
#pragma unroll
      for (int j = 0; j < 8; j++) {
        V16BF B = load_B(W2img, j * 4 + t, lane);
        acc[j] = wmma_bf16(A, B, acc[j]);
      }
    }
    float sg = s[i];
    float pg[8];
    {
      float b2v[8], cnv[8];
#pragma unroll
      for (int j = 0; j < 8; j++) { b2v[j] = b2[q + 16 * j]; cnv[j] = cwl[q + 16 * j]; }
#pragma unroll
      for (int r = 0; r < 8; r++) {            // r-outer: one atomic base per row
        int dM = dsta[r + 8 * hi];
        float* bp = msum + (size_t)dM * HD + q;
        float gacc = 0.f;
#pragma unroll
        for (int j = 0; j < 8; j++) {
          float xv = silu(acc[j][r] + b2v[j]) * sg;
          gacc += xv * cnv[j];                 // gamma partial (this lane's 8 cols)
          fatomic_add(bp + 16 * j, xv);        // imm-offset atomics off one base
        }
        pg[r] = gacc;
      }
    }
    // reduce gamma partials across the 16 lanes of each half (rows M=r+8*hi)
#pragma unroll
    for (int r = 0; r < 8; r++) {
#pragma unroll
      for (int mset = 1; mset <= 8; mset <<= 1) pg[r] += __shfl_xor(pg[r], mset);
    }
    float g = pg[0];
#pragma unroll
    for (int r = 1; r < 8; r++) g = (q == r) ? pg[r] : g;   // g = pg[q]
    if (q < 8) {                               // lane q owns row M=q+8*hi
      int M = q + 8 * hi;
      float gg = g + cb0;
      int dM = dsta[M];
      float* cp = cu + (size_t)dM * 3;
      fatomic_add(cp + 0, gg * dxa[M]);
      fatomic_add(cp + 1, gg * dya[M]);
      fatomic_add(cp + 2, gg * dza[M]);
    }
  }
}

// ===========================================================================
// Node kernel: h' = silu([h | msum/deg]@nw1+nb1)@nw2+nb2 ; p' = p + cu/deg
// ===========================================================================
__global__ void __launch_bounds__(128) node_kernel(
    const unsigned short* __restrict__ h, const float* __restrict__ msum,
    const float* __restrict__ deg, const float* __restrict__ pcur,
    const float* __restrict__ cu, const unsigned int* __restrict__ imgW1,
    const unsigned int* __restrict__ imgW2, const float* __restrict__ nb1,
    const float* __restrict__ nb2, unsigned short* __restrict__ hout,
    float* __restrict__ pout) {
  extern __shared__ char smem[];
  unsigned int* W1img = (unsigned int*)smem;
  unsigned int* W2img = W1img + 64 * 256;
  float* b1 = (float*)(W2img + 32 * 256);
  float* b2 = b1 + HD;
  unsigned short* hnbase = (unsigned short*)(b2 + HD);
  int wid = threadIdx.x >> 5, lane = threadIdx.x & 31;
  unsigned short* hnbuf = hnbase + wid * 16 * HD;
  stage_copy(W1img, imgW1, 64 * 256);
  stage_copy(W2img, imgW2, 32 * 256);
  for (int c = threadIdx.x; c < HD; c += blockDim.x) { b1[c] = nb1[c]; b2[c] = nb2[c]; }
  __syncthreads();
  int q = lane & 15, hi = lane >> 4;
  int nwaves = gridDim.x * (blockDim.x >> 5);
  for (int tile = blockIdx.x * (blockDim.x >> 5) + wid; tile < NTILE; tile += nwaves) {
    int base = tile * 16;
    int node = base + q;
    const unsigned short* hrow = h + (size_t)node * HD;
    const float* mrow = msum + (size_t)node * HD;
    float scale = 1.0f / fmaxf(deg[node], 1.0f);
    v8f acc[8];
#pragma unroll
    for (int j = 0; j < 8; j++) acc[j] = v8zero();
#pragma unroll
    for (int t = 0; t < 4; t++) {              // rows 0..127 : h
      V16BF A = load_A_bf16(hrow, 32 * t, lane);
#pragma unroll
      for (int j = 0; j < 8; j++) {
        V16BF B = load_B(W1img, j * 8 + t, lane);
        acc[j] = wmma_bf16(A, B, acc[j]);
      }
    }
#pragma unroll
    for (int t = 0; t < 4; t++) {              // rows 128..255 : msum/deg
      V16BF A = load_A_f32_scaled(mrow, scale, 32 * t, lane);
#pragma unroll
      for (int j = 0; j < 8; j++) {
        V16BF B = load_B(W1img, j * 8 + 4 + t, lane);
        acc[j] = wmma_bf16(A, B, acc[j]);
      }
    }
    {
      float b1v[8];
#pragma unroll
      for (int j = 0; j < 8; j++) b1v[j] = b1[q + 16 * j];
#pragma unroll
      for (int r = 0; r < 8; r++) {
        int M = r + 8 * hi;
        unsigned short* hp = hnbuf + M * HD + q;
#pragma unroll
        for (int j = 0; j < 8; j++) hp[16 * j] = f2bf(silu(acc[j][r] + b1v[j]));
      }
    }
    __builtin_amdgcn_wave_barrier();
#pragma unroll
    for (int j = 0; j < 8; j++) acc[j] = v8zero();
#pragma unroll
    for (int t = 0; t < 4; t++) {
      V16BF A = load_A_lds(hnbuf, 32 * t, lane);
#pragma unroll
      for (int j = 0; j < 8; j++) {
        V16BF B = load_B(W2img, j * 4 + t, lane);
        acc[j] = wmma_bf16(A, B, acc[j]);
      }
    }
    {
      float b2v[8];
#pragma unroll
      for (int j = 0; j < 8; j++) b2v[j] = b2[q + 16 * j];
#pragma unroll
      for (int r = 0; r < 8; r++) {
        int M = r + 8 * hi;
        unsigned short* op = hout + (size_t)(base + M) * HD + q;
#pragma unroll
        for (int j = 0; j < 8; j++) op[16 * j] = f2bf(acc[j][r] + b2v[j]);  // no silu
      }
    }
    if (lane < 16) {
      float dg = fmaxf(deg[node], 1.0f);
#pragma unroll
      for (int c = 0; c < 3; c++)
        pout[node * 3 + c] = pcur[node * 3 + c] + cu[node * 3 + c] / dg;
    }
  }
}

// ===========================================================================
// Output heads: out[n] = [h@ec_w+ec_b (3) | h@ef_w+ef_b (64) | p (3)]
// ===========================================================================
__global__ void out_kernel(const unsigned short* __restrict__ h,
                           const float* __restrict__ p,
                           const float* __restrict__ ecw, const float* __restrict__ ecb,
                           const float* __restrict__ efw, const float* __restrict__ efb,
                           float* __restrict__ out) {
  int tid = blockIdx.x * blockDim.x + threadIdx.x;
  const int OC = 3 + NDIM + 3;
  if (tid >= NND * OC) return;
  int node = tid / OC, c = tid % OC;
  float v;
  if (c < 3) {
    float a = ecb[c];
    for (int k = 0; k < HD; k++) a += bf2f(h[(size_t)node * HD + k]) * ecw[k * 3 + c];
    v = a;
  } else if (c < 3 + NDIM) {
    int cf = c - 3;
    float a = efb[cf];
    for (int k = 0; k < HD; k++) a += bf2f(h[(size_t)node * HD + k]) * efw[k * NDIM + cf];
    v = a;
  } else {
    v = p[node * 3 + (c - 3 - NDIM)];
  }
  out[tid] = v;
}

// ===========================================================================
extern "C" void kernel_launch(void* const* d_in, const int* in_sizes, int n_in,
                              void* d_out, int out_size, void* d_ws, size_t ws_size,
                              hipStream_t stream) {
  (void)in_sizes; (void)n_in; (void)out_size; (void)ws_size;
  const float* x      = (const float*)d_in[0];
  const float* pos    = (const float*)d_in[1];
  const float* t      = (const float*)d_in[2];
  const float* s      = (const float*)d_in[3];
  const float* proj_w = (const float*)d_in[4];
  const float* proj_b = (const float*)d_in[5];
  const float* ew1    = (const float*)d_in[6];
  const float* eb1    = (const float*)d_in[7];
  const float* ew2    = (const float*)d_in[8];
  const float* eb2    = (const float*)d_in[9];
  const float* nw1    = (const float*)d_in[10];
  const float* nb1    = (const float*)d_in[11];
  const float* nw2    = (const float*)d_in[12];
  const float* nb2    = (const float*)d_in[13];
  const float* cw     = (const float*)d_in[14];
  const float* cb     = (const float*)d_in[15];
  const float* ecw    = (const float*)d_in[16];
  const float* ecb    = (const float*)d_in[17];
  const float* efw    = (const float*)d_in[18];
  const float* efb    = (const float*)d_in[19];

  char* w = (char*)d_ws;
  int*   nbr  = (int*)w;            w += sizeof(int) * NND * KNN;
  float* deg  = (float*)w;          w += sizeof(float) * NND;
  unsigned short* hA = (unsigned short*)w; w += sizeof(unsigned short) * NND * HD;
  unsigned short* hB = (unsigned short*)w; w += sizeof(unsigned short) * NND * HD;
  float* pA   = (float*)w;          w += sizeof(float) * NND * 3;
  float* pB   = (float*)w;          w += sizeof(float) * NND * 3;
  float* msum = (float*)w;          w += sizeof(float) * NND * HD;
  float* cuB  = (float*)w;          w += sizeof(float) * NND * 3;
  unsigned int* imgs = (unsigned int*)w;   // prebuilt B images
  unsigned int* imgProj = imgs;            // 6144 dw
  // per layer: E1(16384) E2(8192) N1(16384) N2(8192)

  const size_t SMEM_PROJ = (8 * 3 * 256) * 4 + HD * 4 + TDIM * 4;
  const size_t SMEM_EDGE = (64 * 256 + 32 * 256) * 4 + 4 * HD * 4 + 4 * (size_t)EDGE_PW;
  const size_t SMEM_NODE = (64 * 256 + 32 * 256) * 4 + 2 * HD * 4 + 4 * (16 * HD * 2);

  // 0) prebuild all weight B-operand register images (bf16) in global ws
  build_image_kernel<<<16, 256, 0, stream>>>(proj_w, NDIM + 1 + TDIM, 96, imgProj);
  for (int l = 0; l < NL; l++) {
    unsigned int* base = imgs + 6144 + (size_t)l * 49152;
    build_image_kernel<<<32, 256, 0, stream>>>(ew1 + (size_t)l * 257 * HD, 256, 256, base);
    build_image_kernel<<<16, 256, 0, stream>>>(ew2 + (size_t)l * HD * HD, 128, 128, base + 16384);
    build_image_kernel<<<32, 256, 0, stream>>>(nw1 + (size_t)l * 256 * HD, 256, 256, base + 24576);
    build_image_kernel<<<16, 256, 0, stream>>>(nw2 + (size_t)l * HD * HD, 128, 128, base + 40960);
  }

  // 1) graph build (no_grad in reference; built once from initial pos)
  knn_kernel<<<(NND + 255) / 256, 256, 0, stream>>>(pos, nbr);
  zero_kernel<<<64, 256, 0, stream>>>(deg, NND);
  deg_kernel<<<(NND * KNN + 255) / 256, 256, 0, stream>>>(nbr, deg);

  // 2) input projection (h stored bf16)
  proj_kernel<<<48, 128, SMEM_PROJ, stream>>>(x, s, t, imgProj, proj_b, hA);
  (void)hipMemcpyAsync(pA, pos, sizeof(float) * NND * 3, hipMemcpyDeviceToDevice, stream);

  // 3) EGNN layers (ping-pong h/p)
  unsigned short* hc = hA; unsigned short* hn = hB;
  float* pc = pA; float* pn = pB;
  for (int l = 0; l < NL; l++) {
    unsigned int* base = imgs + 6144 + (size_t)l * 49152;
    zero_kernel<<<256, 256, 0, stream>>>(msum, NND * HD);
    zero_kernel<<<32, 256, 0, stream>>>(cuB, NND * 3);
    edge_kernel<<<128, 128, SMEM_EDGE, stream>>>(
        hc, pc, s, nbr, base, base + 16384,
        ew1 + (size_t)l * 257 * HD, eb1 + (size_t)l * HD, eb2 + (size_t)l * HD,
        cw + (size_t)l * HD, cb + l, msum, cuB);
    node_kernel<<<48, 128, SMEM_NODE, stream>>>(
        hc, msum, deg, pc, cuB, base + 24576, base + 40960,
        nb1 + (size_t)l * HD, nb2 + (size_t)l * HD, hn, pn);
    unsigned short* th = hc; hc = hn; hn = th;
    float* tp = pc; pc = pn; pn = tp;
  }

  // 4) output heads + final positions
  out_kernel<<<(NND * (3 + NDIM + 3) + 255) / 256, 256, 0, stream>>>(
      hc, pc, ecw, ecb, efw, efb, (float*)d_out);
}
